// InfrastructureGNN_11192684773376
// MI455X (gfx1250) — matrix operand order repaired and checked by reading, not compiled, not run on hardware
//
#include <hip/hip_runtime.h>
#include <hip/hip_bf16.h>

#define NNODES 50000
#define NEDGES 800000
#define NGRAPH 50
#define FIN    15
#define CH     64
#define NH     4
#define HC     256
#define CDIV(a,b) (((a)+(b)-1)/(b))

typedef __attribute__((ext_vector_type(16))) _Float16 v16h;
typedef __attribute__((ext_vector_type(8)))  _Float16 v8h;
typedef __attribute__((ext_vector_type(8)))  float    v8f;

#define CAT16(lo, hi) __builtin_shufflevector((lo), (hi), 0,1,2,3,4,5,6,7,8,9,10,11,12,13,14,15)

// ---------------------------------------------------------------------------
// float atomic max via CAS (guaranteed-compiling, monotonic)
// ---------------------------------------------------------------------------
__device__ __forceinline__ void atomicMaxF32(float* addr, float val) {
  unsigned int* ua = (unsigned int*)addr;
  unsigned int old = *ua;
  while (__uint_as_float(old) < val) {
    unsigned int assumed = old;
    old = atomicCAS(ua, assumed, __float_as_uint(val));
    if (old == assumed) break;
  }
}

// ---------------------------------------------------------------------------
// Fill kernel
// ---------------------------------------------------------------------------
__global__ void gnn_fill(float* p, float val, size_t n) {
  size_t i = (size_t)blockIdx.x * blockDim.x + threadIdx.x;
  if (i < n) p[i] = val;
}

// ---------------------------------------------------------------------------
// Operand packing: A[M,K] f32 -> Ah[M,Kpad] f16 (zero-padded K)
//                  W[K,Nc] f32 -> Wt[Ncpad,Kpad] f16 (transposed, zero-padded)
// ---------------------------------------------------------------------------
__global__ void gnn_pack_a(const float* __restrict__ A, _Float16* __restrict__ Ah,
                           int M, int K, int Kpad) {
  size_t i = (size_t)blockIdx.x * blockDim.x + threadIdx.x;
  if (i >= (size_t)M * Kpad) return;
  int r = (int)(i / Kpad), k = (int)(i % Kpad);
  Ah[i] = (k < K) ? (_Float16)A[(size_t)r * K + k] : (_Float16)0.f;
}

__global__ void gnn_pack_w(const float* __restrict__ W, _Float16* __restrict__ Wt,
                           int K, int Nc, int Kpad, int Ncpad) {
  int i = blockIdx.x * blockDim.x + threadIdx.x;
  if (i >= Ncpad * Kpad) return;
  int c = i / Kpad, k = i % Kpad;
  Wt[i] = (c < Nc && k < K) ? (_Float16)W[(size_t)k * Nc + c] : (_Float16)0.f;
}

// ---------------------------------------------------------------------------
// WMMA GEMM: C[M,Nc] = Ah[M,Kpad](f16) x Wt[Ncpad,Kpad](f16, col-major K runs)
// Block: 256 thr = 8 waves (2 row x 4 col). Block tile 64x128.
// Wave tile 32x32 = 2x2 of 16x16 -> 4 v_wmma per K-step.
// A staged in LDS via b128 copies; B fragments loaded straight from L2-resident
// packed Wt (two 16B runs per lane per fragment).
// ---------------------------------------------------------------------------
__global__ __launch_bounds__(256)
void gnn_wmma_gemm(const _Float16* __restrict__ Ah, const _Float16* __restrict__ Wt,
                   float* __restrict__ C, int M, int Kpad, int Nc) {
  __shared__ _Float16 lA[64][48];   // 64 rows x 32 K halfs (stride 48 = 96B, 16B-aligned)

  const int tid  = threadIdx.x;
  const int lane = tid & 31;
  const int wave = tid >> 5;
  const int rowBase = blockIdx.y * 64;
  const int colBase = blockIdx.x * 128;
  const int wr = (wave >> 2) * 32;   // wave row offset: 0 | 32
  const int wc = (wave & 3) * 32;    // wave col offset: 0 | 32 | 64 | 96
  const int m    = lane & 15;        // row-in-tile (A) / col-in-tile (B,D)
  const int half = lane >> 4;        // lane half per ISA 7.12.2
  const int mofs = half * 8;         // D-matrix M offset per lane half

  // staging assignment: each thread owns one 16B chunk of the 64x32 A tile
  const int arow   = tid >> 2;          // 0..63
  const int achunk = (tid & 3) * 8;     // half-offset 0,8,16,24
  const int gr_l   = rowBase + arow;
  const _Float16* aptr = Ah + (size_t)gr_l * Kpad + achunk;

  v8f acc[2][2] = {};

  for (int k0 = 0; k0 < Kpad; k0 += 32) {
    // ---- stage A tile: one global_load_b128 + one ds_store_b128 per thread
    v8h av = {};
    if (gr_l < M) {
      av = *(const v8h*)(aptr + k0);
      if (k0 + 32 < Kpad) __builtin_prefetch(aptr + k0 + 32, 0, 0);
    }
    *(v8h*)&lA[arow][achunk] = av;
    __syncthreads();

    // ---- A fragments (two row tiles): 2x ds_load_b128 each, concat halves
    v16h a0 = CAT16(*(const v8h*)&lA[wr + m][mofs],
                    *(const v8h*)&lA[wr + m][16 + mofs]);
    v16h a1 = CAT16(*(const v8h*)&lA[wr + 16 + m][mofs],
                    *(const v8h*)&lA[wr + 16 + m][16 + mofs]);

    // ---- B fragments (two col tiles): two 16B runs per lane (k 8h..8h+7, 16+8h..16+8h+7) from packed Wt
    const _Float16* b0p = Wt + (size_t)(colBase + wc + m) * Kpad + k0 + half * 8;
    const _Float16* b1p = Wt + (size_t)(colBase + wc + 16 + m) * Kpad + k0 + half * 8;
    v16h b0 = CAT16(*(const v8h*)b0p, *(const v8h*)(b0p + 16));
    v16h b1 = CAT16(*(const v8h*)b1p, *(const v8h*)(b1p + 16));

    acc[0][0] = __builtin_amdgcn_wmma_f32_16x16x32_f16(false, a0, false, b0,
                                                       (short)0, acc[0][0], false, false);
    acc[0][1] = __builtin_amdgcn_wmma_f32_16x16x32_f16(false, a0, false, b1,
                                                       (short)0, acc[0][1], false, false);
    acc[1][0] = __builtin_amdgcn_wmma_f32_16x16x32_f16(false, a1, false, b0,
                                                       (short)0, acc[1][0], false, false);
    acc[1][1] = __builtin_amdgcn_wmma_f32_16x16x32_f16(false, a1, false, b1,
                                                       (short)0, acc[1][1], false, false);
    __syncthreads();
  }

  // ---- store: D layout VGPR r -> M = r + 8*half; N = lane&15
#pragma unroll
  for (int t = 0; t < 2; ++t)
#pragma unroll
    for (int u = 0; u < 2; ++u) {
      int gc = colBase + wc + u * 16 + m;
#pragma unroll
      for (int r = 0; r < 8; ++r) {
        int gr = rowBase + wr + t * 16 + mofs + r;
        if (gr < M && gc < Nc) C[(size_t)gr * Nc + gc] = acc[t][u][r];
      }
    }
}

// ---------------------------------------------------------------------------
// Per-node attention scores: alpha_s/d[n,h] = dot(h[n,h,:], a{S,D}[h,:])
// ---------------------------------------------------------------------------
__global__ void gnn_alpha(const float* __restrict__ h, const float* __restrict__ aS,
                          const float* __restrict__ aD, float* as_out, float* ad_out,
                          int N_, int heads, int Cc) {
  int i = blockIdx.x * blockDim.x + threadIdx.x;
  if (i >= N_ * heads) return;
  int n = i / heads, hh = i % heads;
  const float* hp = h + ((size_t)n * heads + hh) * Cc;
  float s = 0.f, d = 0.f;
  for (int c = 0; c < Cc; ++c) {
    s += hp[c] * aS[hh * Cc + c];
    d += hp[c] * aD[hh * Cc + c];
  }
  as_out[i] = s;
  ad_out[i] = d;
}

// ---------------------------------------------------------------------------
// Edge passes (index >= nE are the self loops s=d=i-nE)
// ---------------------------------------------------------------------------
__global__ void gnn_edge_max(const int* __restrict__ src, const int* __restrict__ dst,
                             const float* __restrict__ as_, const float* __restrict__ ad_,
                             float* segmax, int nE, int total, int heads) {
  int i = blockIdx.x * blockDim.x + threadIdx.x;
  if (i >= total) return;
  int s = (i < nE) ? src[i] : (i - nE);
  int d = (i < nE) ? dst[i] : (i - nE);
  for (int hh = 0; hh < heads; ++hh) {
    float e = as_[s * heads + hh] + ad_[d * heads + hh];
    e = (e > 0.f) ? e : 0.2f * e;                 // leaky relu
    atomicMaxF32(&segmax[d * heads + hh], e);
  }
}

__global__ void gnn_edge_exp(const int* __restrict__ src, const int* __restrict__ dst,
                             const float* __restrict__ as_, const float* __restrict__ ad_,
                             const float* __restrict__ segmax, float* segsum,
                             float* ebuf, int nE, int total, int heads) {
  int i = blockIdx.x * blockDim.x + threadIdx.x;
  if (i >= total) return;
  int s = (i < nE) ? src[i] : (i - nE);
  int d = (i < nE) ? dst[i] : (i - nE);
  for (int hh = 0; hh < heads; ++hh) {
    float e = as_[s * heads + hh] + ad_[d * heads + hh];
    e = (e > 0.f) ? e : 0.2f * e;
    float ex = __expf(e - segmax[d * heads + hh]);
    ebuf[(size_t)i * heads + hh] = ex;
    atomicAdd(&segsum[d * heads + hh], ex);
  }
}

__global__ void gnn_edge_agg(const int* __restrict__ src, const int* __restrict__ dst,
                             const float* __restrict__ hfeat, const float* __restrict__ ebuf,
                             const float* __restrict__ segsum, float* out,
                             int nE, int total, int heads, int Cc) {
  int i = blockIdx.x * blockDim.x + threadIdx.x;
  if (i >= total * heads) return;
  int e = i / heads, hh = i % heads;
  int s = (e < nE) ? src[e] : (e - nE);
  int d = (e < nE) ? dst[e] : (e - nE);
  float alpha = ebuf[(size_t)e * heads + hh] / segsum[d * heads + hh];
  const float* hp = hfeat + ((size_t)s * heads + hh) * Cc;
  float* op = out + ((size_t)d * heads + hh) * Cc;
  for (int c = 0; c < Cc; ++c) atomicAdd(&op[c], hp[c] * alpha);
}

__global__ void gnn_bias_elu(float* buf, const float* __restrict__ bias, int Nc, size_t n) {
  size_t i = (size_t)blockIdx.x * blockDim.x + threadIdx.x;
  if (i >= n) return;
  float v = buf[i] + bias[i % Nc];
  buf[i] = (v > 0.f) ? v : (__expf(v) - 1.f);
}

// ---------------------------------------------------------------------------
// Small dense layer: one thread per output element. act 0=none,1=relu,2=sigmoid
// ---------------------------------------------------------------------------
__global__ void gnn_dense(const float* __restrict__ A, const float* __restrict__ W,
                          const float* __restrict__ bias, float* out,
                          int M, int K, int Nc, int act) {
  int i = blockIdx.x * blockDim.x + threadIdx.x;
  if (i >= M * Nc) return;
  int m = i / Nc, j = i % Nc;
  float s = bias[j];
  const float* ap = A + (size_t)m * K;
  for (int k = 0; k < K; ++k) s += ap[k] * W[k * Nc + j];
  if (act == 1) s = fmaxf(s, 0.f);
  else if (act == 2) s = 1.f / (1.f + __expf(-s));
  out[i] = s;
}

// ---------------------------------------------------------------------------
// Global softmax over all nodes (3 passes)
// ---------------------------------------------------------------------------
__global__ void gnn_gmax(const float* v, float* gmax, int n) {
  int i = blockIdx.x * blockDim.x + threadIdx.x;
  if (i < n) atomicMaxF32(gmax, v[i]);
}
__global__ void gnn_gsumexp(const float* v, const float* gmax, float* gsum, int n) {
  int i = blockIdx.x * blockDim.x + threadIdx.x;
  if (i < n) atomicAdd(gsum, __expf(v[i] - gmax[0]));
}
__global__ void gnn_softmax_write(const float* v, const float* gmax, const float* gsum,
                                  float* out, int n) {
  int i = blockIdx.x * blockDim.x + threadIdx.x;
  if (i < n) out[i] = __expf(v[i] - gmax[0]) / gsum[0];
}

// ---------------------------------------------------------------------------
// Graph pooling (mean + max per graph)
// ---------------------------------------------------------------------------
__global__ void gnn_pool(const float* __restrict__ x3, const int* __restrict__ batch,
                         float* cnt, float* msum, float* mx, int N_, int Cc) {
  int i = blockIdx.x * blockDim.x + threadIdx.x;
  if (i >= N_ * Cc) return;
  int n = i / Cc, c = i % Cc;
  int b = batch[n];
  float v = x3[(size_t)n * Cc + c];
  if (c == 0) atomicAdd(&cnt[b], 1.0f);
  atomicAdd(&msum[b * Cc + c], v);
  atomicMaxF32(&mx[b * Cc + c], v);
}
__global__ void gnn_pool_finalize(const float* cnt, const float* msum, const float* mx,
                                  float* g, int Bn, int Cc) {
  int i = blockIdx.x * blockDim.x + threadIdx.x;
  if (i >= Bn * Cc) return;
  int b = i / Cc, c = i % Cc;
  float ct = cnt[b];
  g[b * 2 * Cc + c]      = msum[b * Cc + c] / fmaxf(ct, 1.0f);
  g[b * 2 * Cc + Cc + c] = (ct > 0.f) ? mx[b * Cc + c] : 0.0f;
}

// ---------------------------------------------------------------------------
// Host launch
// ---------------------------------------------------------------------------
extern "C" void kernel_launch(void* const* d_in, const int* in_sizes, int n_in,
                              void* d_out, int out_size, void* d_ws, size_t ws_size,
                              hipStream_t stream) {
  (void)in_sizes; (void)n_in; (void)out_size; (void)ws_size;

  const float* x    = (const float*)d_in[0];
  const int*   ei   = (const int*)d_in[1];
  const int*   batch= (const int*)d_in[2];
  const float* W1  = (const float*)d_in[3];
  const float* aS1 = (const float*)d_in[4];
  const float* aD1 = (const float*)d_in[5];
  const float* b1  = (const float*)d_in[6];
  const float* W2  = (const float*)d_in[7];
  const float* aS2 = (const float*)d_in[8];
  const float* aD2 = (const float*)d_in[9];
  const float* b2  = (const float*)d_in[10];
  const float* W3  = (const float*)d_in[11];
  const float* aS3 = (const float*)d_in[12];
  const float* aD3 = (const float*)d_in[13];
  const float* b3  = (const float*)d_in[14];
  const float* Wna1= (const float*)d_in[15];
  const float* bna1= (const float*)d_in[16];
  const float* Wna2= (const float*)d_in[17];
  const float* bna2= (const float*)d_in[18];
  const float* Wc1 = (const float*)d_in[19];
  const float* bc1 = (const float*)d_in[20];
  const float* Wc2 = (const float*)d_in[21];
  const float* bc2 = (const float*)d_in[22];
  const float* Wc3 = (const float*)d_in[23];
  const float* bc3 = (const float*)d_in[24];

  const int* src = ei;            // edge_index row 0
  const int* dst = ei + NEDGES;   // edge_index row 1
  float* outp = (float*)d_out;    // [0..B) risk, [B..B+N) node_attention

  // workspace carve-up (256B aligned)
  char* ws = (char*)d_ws;
  size_t off = 0;
  auto takeb = [&](size_t bytes) -> void* {
    void* p = ws + off;
    off += (bytes + 255) & ~(size_t)255;
    return p;
  };
  auto take = [&](size_t nfloats) -> float* { return (float*)takeb(nfloats * 4); };

  float*    big0    = take((size_t)NNODES * HC);        // h (pre-agg features)
  float*    big1    = take((size_t)NNODES * HC);        // aggregated / activations
  _Float16* xh      = (_Float16*)takeb((size_t)NNODES * HC * 2);  // packed f16 A
  _Float16* wt      = (_Float16*)takeb((size_t)HC * HC * 2);      // packed f16 W^T
  float*    f_as    = take((size_t)NNODES * NH);        // also reused as na
  float*    f_ad    = take((size_t)NNODES * NH);
  float*    f_segmax= take((size_t)NNODES * NH);
  float*    f_segsum= take((size_t)NNODES * NH);
  float*    f_ebuf  = take((size_t)(NEDGES + NNODES) * NH);
  float*    f_cnt   = take(NGRAPH);
  float*    f_msum  = take((size_t)NGRAPH * CH);
  float*    f_mx    = take((size_t)NGRAPH * CH);
  float*    f_g     = take((size_t)NGRAPH * 2 * CH);
  float*    f_c1    = take((size_t)NGRAPH * 128);
  float*    f_c2    = take((size_t)NGRAPH * 64);
  float*    f_scal  = take(2);

  const int total = NEDGES + NNODES;
  const float NEGBIG = -1e30f;

  auto run_gat = [&](const float* xin, int Kin, const float* Wm, const float* aSw,
                     const float* aDw, const float* bw, int heads) {
    const int Nc    = heads * CH;
    const int Kpad  = (Kin + 31) & ~31;
    const int Ncpad = (Nc + 127) & ~127;

    gnn_pack_a<<<CDIV((size_t)NNODES * Kpad, 256), 256, 0, stream>>>(xin, xh,
                                                                     NNODES, Kin, Kpad);
    gnn_pack_w<<<CDIV(Ncpad * Kpad, 256), 256, 0, stream>>>(Wm, wt, Kin, Nc,
                                                            Kpad, Ncpad);
    dim3 gb(CDIV(Nc, 128), CDIV(NNODES, 64));
    gnn_wmma_gemm<<<gb, 256, 0, stream>>>(xh, wt, big0, NNODES, Kpad, Nc);

    const int nh = NNODES * heads;
    gnn_alpha<<<CDIV(nh, 256), 256, 0, stream>>>(big0, aSw, aDw, f_as, f_ad,
                                                 NNODES, heads, CH);
    gnn_fill<<<CDIV(nh, 256), 256, 0, stream>>>(f_segmax, NEGBIG, (size_t)nh);
    gnn_fill<<<CDIV(nh, 256), 256, 0, stream>>>(f_segsum, 0.0f, (size_t)nh);
    gnn_fill<<<CDIV((size_t)NNODES * Nc, 256), 256, 0, stream>>>(big1, 0.0f,
                                                                 (size_t)NNODES * Nc);
    gnn_edge_max<<<CDIV(total, 256), 256, 0, stream>>>(src, dst, f_as, f_ad,
                                                       f_segmax, NEDGES, total, heads);
    gnn_edge_exp<<<CDIV(total, 256), 256, 0, stream>>>(src, dst, f_as, f_ad, f_segmax,
                                                       f_segsum, f_ebuf, NEDGES, total, heads);
    gnn_edge_agg<<<CDIV(total * heads, 256), 256, 0, stream>>>(src, dst, big0, f_ebuf,
                                                               f_segsum, big1, NEDGES,
                                                               total, heads, CH);
    gnn_bias_elu<<<CDIV((size_t)NNODES * Nc, 256), 256, 0, stream>>>(big1, bw, Nc,
                                                                     (size_t)NNODES * Nc);
  };

  // ---- GAT layers: x1/x2/x3 end up in big1 ----
  run_gat(x,    FIN, W1, aS1, aD1, b1, NH);
  run_gat(big1, HC,  W2, aS2, aD2, b2, NH);
  run_gat(big1, HC,  W3, aS3, aD3, b3, 1);   // x3 = big1 (N x 64)

  // ---- node attention MLP: 64 -> 32 -> 1, softmax over all N ----
  gnn_dense<<<CDIV(NNODES * 32, 256), 256, 0, stream>>>(big1, Wna1, bna1, big0,
                                                        NNODES, CH, 32, 1);
  gnn_dense<<<CDIV(NNODES, 256), 256, 0, stream>>>(big0, Wna2, bna2, f_as,
                                                   NNODES, 32, 1, 0);
  gnn_fill<<<1, 1, 0, stream>>>(f_scal, NEGBIG, 1);
  gnn_fill<<<1, 1, 0, stream>>>(f_scal + 1, 0.0f, 1);
  gnn_gmax<<<CDIV(NNODES, 256), 256, 0, stream>>>(f_as, f_scal, NNODES);
  gnn_gsumexp<<<CDIV(NNODES, 256), 256, 0, stream>>>(f_as, f_scal, f_scal + 1, NNODES);
  gnn_softmax_write<<<CDIV(NNODES, 256), 256, 0, stream>>>(f_as, f_scal, f_scal + 1,
                                                           outp + NGRAPH, NNODES);

  // ---- graph pooling: mean + max per graph ----
  gnn_fill<<<1, 64, 0, stream>>>(f_cnt, 0.0f, NGRAPH);
  gnn_fill<<<CDIV(NGRAPH * CH, 256), 256, 0, stream>>>(f_msum, 0.0f, (size_t)NGRAPH * CH);
  gnn_fill<<<CDIV(NGRAPH * CH, 256), 256, 0, stream>>>(f_mx, NEGBIG, (size_t)NGRAPH * CH);
  gnn_pool<<<CDIV(NNODES * CH, 256), 256, 0, stream>>>(big1, batch, f_cnt, f_msum, f_mx,
                                                       NNODES, CH);
  gnn_pool_finalize<<<CDIV(NGRAPH * CH, 256), 256, 0, stream>>>(f_cnt, f_msum, f_mx, f_g,
                                                                NGRAPH, CH);

  // ---- classifier: 128 -> 128 -> 64 -> 1 (sigmoid) ----
  gnn_dense<<<CDIV(NGRAPH * 128, 256), 256, 0, stream>>>(f_g, Wc1, bc1, f_c1,
                                                         NGRAPH, 2 * CH, 128, 1);
  gnn_dense<<<CDIV(NGRAPH * 64, 256), 256, 0, stream>>>(f_c1, Wc2, bc2, f_c2,
                                                        NGRAPH, 128, 64, 1);
  gnn_dense<<<1, 64, 0, stream>>>(f_c2, Wc3, bc3, outp, NGRAPH, 64, 1, 2);
}
